// CausalSelfAttention_9775345565790
// MI455X (gfx1250) — matrix operand order repaired and checked by reading, not compiled, not run on hardware
//
#include <hip/hip_runtime.h>
#include <hip/hip_bf16.h>

typedef __attribute__((ext_vector_type(16))) _Float16 v16h;
typedef __attribute__((ext_vector_type(8)))  _Float16 v8h;
typedef __attribute__((ext_vector_type(8)))  float    v8f;
typedef __attribute__((ext_vector_type(4)))  int      int4v;

union Frag16 { v16h f; v8h h[2]; };

// ---- gfx1250 async load-to-LDS path (guarded; falls back to manual copy) ----
#if defined(__has_builtin)
#  if __has_builtin(__builtin_amdgcn_global_load_async_to_lds_b128)
#    define HAVE_ASYNC_LDS 1
#  endif
#endif
#ifndef HAVE_ASYNC_LDS
#  define HAVE_ASYNC_LDS 0
#endif

// builtin signature (from hipcc diagnostic): (int4 AS1*, int4 AS3*, imm, imm)
typedef __attribute__((address_space(1))) int4v* gv4i_t;
typedef __attribute__((address_space(3))) int4v* lv4i_t;

__device__ __forceinline__ void wait_asynccnt0() {
#if defined(__has_builtin) && __has_builtin(__builtin_amdgcn_s_wait_asynccnt)
    __builtin_amdgcn_s_wait_asynccnt(0);
#else
    asm volatile("s_wait_asynccnt 0" ::: "memory");
#endif
}

__device__ __forceinline__ v8f wmma_f16(v16h a, v16h b, v8f c) {
    // D = A(16x32 f16) * B(32x16 f16) + C(16x16 f32)
    return __builtin_amdgcn_wmma_f32_16x16x32_f16(
        /*neg_a=*/false, a, /*neg_b=*/false, b,
        /*c_mod=*/(short)0, c, /*reuse_a=*/false, /*reuse_b=*/false);
}

// ---------------------------------------------------------------------------
// GEMM: C(MxN) = A(MxK) * B(KxN) + bias, via v_wmma_f32_16x16x32_f16.
// Block = 256 threads (8 waves), tile 128x64, BK=32.
// Waves in a 4x2 grid; each wave owns a 32x32 sub-tile (4 WMMA / K-step).
// A staged f32->f16 (or f16 passthrough) to LDS [128][40]; B staged transposed
// to LDS [n][k] = [64][40] so B-fragments are contiguous 16-half reads.
// ---------------------------------------------------------------------------
template<bool A_HALF, bool OUT_HALF>
__global__ __launch_bounds__(256) void gemm_wmma_k(
    const void* __restrict__ Ap, const float* __restrict__ Bp,
    const float* __restrict__ bias, void* __restrict__ Cp,
    int M, int N, int K)
{
    __shared__ __align__(16) _Float16 sa[128][40];
    __shared__ __align__(16) _Float16 sbt[64][40];

    const int tid  = threadIdx.x;
    const int m0   = blockIdx.y * 128;
    const int n0   = blockIdx.x * 64;
    const int w    = tid >> 5;          // wave 0..7
    const int lane = tid & 31;
    const int l16  = lane & 15;
    const int hf   = lane >> 4;         // half-wave 0/1
    const int mw   = (w & 3) * 32;      // wave row base (2 m-tiles)
    const int nh   = (w >> 2) * 32;     // wave col base (2 n-tiles)

    v8f acc00 = {}, acc01 = {}, acc10 = {}, acc11 = {};

    const int ar = tid >> 1,  ac = (tid & 1) * 16;  // A tile: 128x32
    const int br = tid >> 3,  bc = (tid & 7) * 8;   // B tile: 32x64

    for (int k0 = 0; k0 < K; k0 += 32) {
        // prefetch next K-step tiles into cache (global_prefetch_b8)
        if (k0 + 32 < K) {
            if (A_HALF)
                __builtin_prefetch((const _Float16*)Ap + (size_t)(m0 + ar) * K + k0 + 32 + ac, 0, 1);
            else
                __builtin_prefetch((const float*)Ap + (size_t)(m0 + ar) * K + k0 + 32 + ac, 0, 1);
            __builtin_prefetch(Bp + (size_t)(k0 + 32 + br) * N + n0 + bc, 0, 1);
        }

        __syncthreads();   // guard LDS reuse from previous iteration

        // --- stage A tile (convert to f16): 16 elements per thread ---
        if (A_HALF) {
            const _Float16* A = (const _Float16*)Ap + (size_t)(m0 + ar) * K + k0 + ac;
            v8h a0 = *(const v8h*)A;
            v8h a1 = *(const v8h*)(A + 8);
            *(v8h*)&sa[ar][ac]     = a0;
            *(v8h*)&sa[ar][ac + 8] = a1;
        } else {
            const float* A = (const float*)Ap + (size_t)(m0 + ar) * K + k0 + ac;
            const float4* af = (const float4*)A;
            float4 f0 = af[0], f1 = af[1], f2 = af[2], f3 = af[3];
            v8h a0, a1;
            a0[0]=(_Float16)f0.x; a0[1]=(_Float16)f0.y; a0[2]=(_Float16)f0.z; a0[3]=(_Float16)f0.w;
            a0[4]=(_Float16)f1.x; a0[5]=(_Float16)f1.y; a0[6]=(_Float16)f1.z; a0[7]=(_Float16)f1.w;
            a1[0]=(_Float16)f2.x; a1[1]=(_Float16)f2.y; a1[2]=(_Float16)f2.z; a1[3]=(_Float16)f2.w;
            a1[4]=(_Float16)f3.x; a1[5]=(_Float16)f3.y; a1[6]=(_Float16)f3.z; a1[7]=(_Float16)f3.w;
            *(v8h*)&sa[ar][ac]     = a0;
            *(v8h*)&sa[ar][ac + 8] = a1;
        }

        // --- stage B tile transposed: sbt[n][k] ---
        {
            const float4* bf = (const float4*)(Bp + (size_t)(k0 + br) * N + n0 + bc);
            float4 g0 = bf[0], g1 = bf[1];
            sbt[bc+0][br] = (_Float16)g0.x; sbt[bc+1][br] = (_Float16)g0.y;
            sbt[bc+2][br] = (_Float16)g0.z; sbt[bc+3][br] = (_Float16)g0.w;
            sbt[bc+4][br] = (_Float16)g1.x; sbt[bc+5][br] = (_Float16)g1.y;
            sbt[bc+6][br] = (_Float16)g1.z; sbt[bc+7][br] = (_Float16)g1.w;
        }
        __syncthreads();

        // --- fragments (contiguous k: lane half h holds k = 16h..16h+15) ---
        Frag16 af0, af1, bf0, bf1;
        const int kb8 = hf * 8, ko = hf * 16;
        af0.h[0] = *(const v8h*)&sa[mw + l16][ko];
        af0.h[1] = *(const v8h*)&sa[mw + l16][ko + 8];
        af1.h[0] = *(const v8h*)&sa[mw + 16 + l16][ko];
        af1.h[1] = *(const v8h*)&sa[mw + 16 + l16][ko + 8];

        bf0.h[0] = *(const v8h*)&sbt[nh + l16][ko];
        bf0.h[1] = *(const v8h*)&sbt[nh + l16][ko + 8];
        bf1.h[0] = *(const v8h*)&sbt[nh + 16 + l16][ko];
        bf1.h[1] = *(const v8h*)&sbt[nh + 16 + l16][ko + 8];

        acc00 = wmma_f16(af0.f, bf0.f, acc00);
        acc01 = wmma_f16(af0.f, bf1.f, acc01);
        acc10 = wmma_f16(af1.f, bf0.f, acc10);
        acc11 = wmma_f16(af1.f, bf1.f, acc11);
    }

    // --- epilogue: bias + store ---
    const int ncol0 = n0 + nh + l16, ncol1 = ncol0 + 16;
    const float bv0 = bias[ncol0], bv1 = bias[ncol1];
    #pragma unroll
    for (int r = 0; r < 8; ++r) {
        const int mrow0 = m0 + mw + r + hf * 8;
        const int mrow1 = mrow0 + 16;
        const float v00 = acc00[r] + bv0, v01 = acc01[r] + bv1;
        const float v10 = acc10[r] + bv0, v11 = acc11[r] + bv1;
        if (OUT_HALF) {
            _Float16* C = (_Float16*)Cp;
            C[(size_t)mrow0 * N + ncol0] = (_Float16)v00;
            C[(size_t)mrow0 * N + ncol1] = (_Float16)v01;
            C[(size_t)mrow1 * N + ncol0] = (_Float16)v10;
            C[(size_t)mrow1 * N + ncol1] = (_Float16)v11;
        } else {
            float* C = (float*)Cp;
            C[(size_t)mrow0 * N + ncol0] = v00;
            C[(size_t)mrow0 * N + ncol1] = v01;
            C[(size_t)mrow1 * N + ncol0] = v10;
            C[(size_t)mrow1 * N + ncol1] = v11;
        }
    }
}

// ---------------------------------------------------------------------------
// RoPE (reference's interleaved-variant) + head-major scatter.
// One thread per (b,t,h,i), i in 0..31.  q gets the 1/sqrt(64) scale folded in.
// ---------------------------------------------------------------------------
__global__ __launch_bounds__(256) void rope_split_k(
    const _Float16* __restrict__ qkv,
    _Float16* __restrict__ qb, _Float16* __restrict__ kb, _Float16* __restrict__ vb)
{
    const int idx = blockIdx.x * blockDim.x + threadIdx.x;  // < 2^21
    const int i = idx & 31;
    const int h = (idx >> 5) & 15;
    const int t = (idx >> 9) & 2047;
    const int b = idx >> 20;

    const _Float16* row = qkv + (size_t)(b * 2048 + t) * 3072;
    const float tf = (float)t;

    // reference: rope = concat(sin(f), cos(f)); sin_u = rope[2i], cos_u = rope[2i+1]
    int p0, p1; bool hi;
    if (i < 16) { p0 = 2 * i;      p1 = 2 * i + 1;  hi = false; }
    else        { p0 = 2 * i - 32; p1 = 2 * i - 31; hi = true;  }
    const float c = -13.287712379549449f / 64.0f;   // -log2(10000)/64
    const float f0 = tf * exp2f((float)(2 * p0) * c);
    const float f1 = tf * exp2f((float)(2 * p1) * c);
    const float sin_u = hi ? __cosf(f0) : __sinf(f0);
    const float cos_u = hi ? __cosf(f1) : __sinf(f1);

    const size_t dbase = ((size_t)(b * 16 + h) * 2048 + t) * 64;
    const int src = h * 64 + 2 * i;

    {   // q (scaled by HD^-0.5 = 0.125)
        float x1 = (float)row[src], x2 = (float)row[src + 1];
        qb[dbase + i]      = (_Float16)(0.125f * (x1 * cos_u - x2 * sin_u));
        qb[dbase + 32 + i] = (_Float16)(0.125f * (x1 * sin_u + x2 * cos_u));
    }
    {   // k
        float x1 = (float)row[1024 + src], x2 = (float)row[1024 + src + 1];
        kb[dbase + i]      = (_Float16)(x1 * cos_u - x2 * sin_u);
        kb[dbase + 32 + i] = (_Float16)(x1 * sin_u + x2 * cos_u);
    }
    // v (straight copy)
    vb[dbase + 2 * i]     = row[2048 + src];
    vb[dbase + 2 * i + 1] = row[2048 + src + 1];
}

// ---------------------------------------------------------------------------
// Flash-attention forward: block = (one head, 64 query rows), 4 waves x 16 rows.
// Streams 64-key tiles (16 WMMA per wave per tile); K tile staged via
// GLOBAL_LOAD_ASYNC_TO_LDS when available; V staged with an LDS transpose so
// p@V B-fragments are contiguous.  Online softmax uses __shfl_xor reductions
// across the 16-lane row groups (wave32).
// ---------------------------------------------------------------------------
__global__ __launch_bounds__(128) void attn_fwd_k(
    const _Float16* __restrict__ qb, const _Float16* __restrict__ kb,
    const _Float16* __restrict__ vb, _Float16* __restrict__ ob)
{
    __shared__ __align__(16) _Float16 sq[64][72];      // q tile  [row][hd]
    __shared__ __align__(16) _Float16 sk[64][72];      // k tile  [key][hd]
    __shared__ __align__(16) _Float16 svt[64][72];     // v tile  [hd][key] (transposed)
    __shared__ __align__(16) _Float16 sp[4][16][72];   // per-wave P scratch [row][key]

    const int qt  = blockIdx.x;         // query tile (64 rows)
    const int bh  = blockIdx.y;         // b*16+h
    const int tid = threadIdx.x;
    const int wv  = tid >> 5;
    const int lane = tid & 31;
    const int l16 = lane & 15;
    const int hf  = lane >> 4;

    const size_t hoff = (size_t)bh * 2048 * 64;
    const int srow = tid >> 1, scol = (tid & 1) * 32;   // 64x64 staging coords

    // --- stage q tile ---
    {
        const v8h* src = (const v8h*)(qb + hoff + (size_t)(qt * 64 + srow) * 64 + scol);
        v8h t0 = src[0], t1 = src[1], t2 = src[2], t3 = src[3];
        *(v8h*)&sq[srow][scol]      = t0;
        *(v8h*)&sq[srow][scol + 8]  = t1;
        *(v8h*)&sq[srow][scol + 16] = t2;
        *(v8h*)&sq[srow][scol + 24] = t3;
    }
    __syncthreads();

    // q A-fragments for the two 32-wide hd chunks (held in registers all loop)
    Frag16 qf0, qf1;
    const int mA = l16, kb8 = hf * 8, ko = hf * 16;
    {
        const int mr = wv * 16 + mA;
        qf0.h[0] = *(const v8h*)&sq[mr][ko];
        qf0.h[1] = *(const v8h*)&sq[mr][ko + 8];
        qf1.h[0] = *(const v8h*)&sq[mr][32 + ko];
        qf1.h[1] = *(const v8h*)&sq[mr][32 + ko + 8];
    }

    v8f acc0 = {}, acc1 = {}, acc2 = {}, acc3 = {};
    float mi[8], li[8];
    #pragma unroll
    for (int r = 0; r < 8; ++r) { mi[r] = -1e30f; li[r] = 0.0f; }

    const int nkt = qt + 1;             // 64-key tiles, causal bound

    for (int kt = 0; kt < nkt; ++kt) {
        __syncthreads();    // previous tile fully consumed

        // --- stage K tile [key][hd] (async DMA to LDS when available) ---
        {
            const _Float16* gk = kb + hoff + (size_t)(kt * 64 + srow) * 64 + scol;
            _Float16* lk = &sk[srow][scol];
#if HAVE_ASYNC_LDS
            __builtin_amdgcn_global_load_async_to_lds_b128((gv4i_t)gk, (lv4i_t)lk, 0,  0);
            __builtin_amdgcn_global_load_async_to_lds_b128((gv4i_t)gk, (lv4i_t)lk, 16, 0);
            __builtin_amdgcn_global_load_async_to_lds_b128((gv4i_t)gk, (lv4i_t)lk, 32, 0);
            __builtin_amdgcn_global_load_async_to_lds_b128((gv4i_t)gk, (lv4i_t)lk, 48, 0);
#else
            const v8h* ks = (const v8h*)gk;
            v8h k0 = ks[0], k1 = ks[1], k2 = ks[2], k3 = ks[3];
            *(v8h*)(lk)      = k0;
            *(v8h*)(lk + 8)  = k1;
            *(v8h*)(lk + 16) = k2;
            *(v8h*)(lk + 24) = k3;
#endif
        }
        // --- stage V tile transposed: svt[hd][key] ---
        {
            const v8h* vs = (const v8h*)(vb + hoff + (size_t)(kt * 64 + srow) * 64 + scol);
            v8h v0 = vs[0], v1 = vs[1], v2 = vs[2], v3 = vs[3];
            #pragma unroll
            for (int j = 0; j < 8; ++j) {
                svt[scol + j][srow]      = v0[j];
                svt[scol + 8 + j][srow]  = v1[j];
                svt[scol + 16 + j][srow] = v2[j];
                svt[scol + 24 + j][srow] = v3[j];
            }
        }
#if HAVE_ASYNC_LDS
        wait_asynccnt0();
#endif
        __syncthreads();

        // --- S = q @ k^T  (four 16-key column tiles, two hd chunks each) ---
        v8f s[4];
        #pragma unroll
        for (int ct = 0; ct < 4; ++ct) {
            Frag16 bf;
            const int n = ct * 16 + l16;            // key within tile
            bf.h[0] = *(const v8h*)&sk[n][ko];      // hd chunk 0
            bf.h[1] = *(const v8h*)&sk[n][ko + 8];
            v8f t = {};
            t = wmma_f16(qf0.f, bf.f, t);
            bf.h[0] = *(const v8h*)&sk[n][32 + ko]; // hd chunk 1
            bf.h[1] = *(const v8h*)&sk[n][32 + ko + 8];
            s[ct] = wmma_f16(qf1.f, bf.f, t);
        }

        // --- causal mask + online softmax (rows live in 16-lane groups) ---
        #pragma unroll
        for (int r = 0; r < 8; ++r) {
            const int qg = qt * 64 + wv * 16 + r + hf * 8;
            const int kg = kt * 64 + l16;
            float a0 = (kg      <= qg) ? s[0][r] : -1e30f;
            float a1 = (kg + 16 <= qg) ? s[1][r] : -1e30f;
            float a2 = (kg + 32 <= qg) ? s[2][r] : -1e30f;
            float a3 = (kg + 48 <= qg) ? s[3][r] : -1e30f;
            float rm = fmaxf(fmaxf(a0, a1), fmaxf(a2, a3));
            #pragma unroll
            for (int off = 1; off < 16; off <<= 1)
                rm = fmaxf(rm, __shfl_xor(rm, off, 32));
            const float mn = fmaxf(mi[r], rm);
            const float al = __expf(mi[r] - mn);
            const float p0 = __expf(a0 - mn);
            const float p1 = __expf(a1 - mn);
            const float p2 = __expf(a2 - mn);
            const float p3 = __expf(a3 - mn);
            float rs = (p0 + p1) + (p2 + p3);
            #pragma unroll
            for (int off = 1; off < 16; off <<= 1)
                rs += __shfl_xor(rs, off, 32);
            li[r] = li[r] * al + rs;
            mi[r] = mn;
            acc0[r] *= al; acc1[r] *= al; acc2[r] *= al; acc3[r] *= al;
            const int pr = r + hf * 8;
            sp[wv][pr][l16]      = (_Float16)p0;
            sp[wv][pr][16 + l16] = (_Float16)p1;
            sp[wv][pr][32 + l16] = (_Float16)p2;
            sp[wv][pr][48 + l16] = (_Float16)p3;
        }
        asm volatile("s_wait_dscnt 0" ::: "memory");

        // --- P into A-fragment layout (wave-private LDS round trip) ---
        Frag16 pf0, pf1;
        pf0.h[0] = *(const v8h*)&sp[wv][mA][ko];        // keys 0..31
        pf0.h[1] = *(const v8h*)&sp[wv][mA][ko + 8];
        pf1.h[0] = *(const v8h*)&sp[wv][mA][32 + ko];   // keys 32..63
        pf1.h[1] = *(const v8h*)&sp[wv][mA][32 + ko + 8];

        // --- O += P @ V  (four 16-hd column tiles x two 32-key chunks) ---
        {
            Frag16 vf;
            vf.h[0] = *(const v8h*)&svt[l16][ko];
            vf.h[1] = *(const v8h*)&svt[l16][ko + 8];
            acc0 = wmma_f16(pf0.f, vf.f, acc0);
            vf.h[0] = *(const v8h*)&svt[l16][32 + ko];
            vf.h[1] = *(const v8h*)&svt[l16][32 + ko + 8];
            acc0 = wmma_f16(pf1.f, vf.f, acc0);

            vf.h[0] = *(const v8h*)&svt[16 + l16][ko];
            vf.h[1] = *(const v8h*)&svt[16 + l16][ko + 8];
            acc1 = wmma_f16(pf0.f, vf.f, acc1);
            vf.h[0] = *(const v8h*)&svt[16 + l16][32 + ko];
            vf.h[1] = *(const v8h*)&svt[16 + l16][32 + ko + 8];
            acc1 = wmma_f16(pf1.f, vf.f, acc1);

            vf.h[0] = *(const v8h*)&svt[32 + l16][ko];
            vf.h[1] = *(const v8h*)&svt[32 + l16][ko + 8];
            acc2 = wmma_f16(pf0.f, vf.f, acc2);
            vf.h[0] = *(const v8h*)&svt[32 + l16][32 + ko];
            vf.h[1] = *(const v8h*)&svt[32 + l16][32 + ko + 8];
            acc2 = wmma_f16(pf1.f, vf.f, acc2);

            vf.h[0] = *(const v8h*)&svt[48 + l16][ko];
            vf.h[1] = *(const v8h*)&svt[48 + l16][ko + 8];
            acc3 = wmma_f16(pf0.f, vf.f, acc3);
            vf.h[0] = *(const v8h*)&svt[48 + l16][32 + ko];
            vf.h[1] = *(const v8h*)&svt[48 + l16][32 + ko + 8];
            acc3 = wmma_f16(pf1.f, vf.f, acc3);
        }
    }

    // --- normalize + store to [B*T][D] f16 (head-interleaved) ---
    const int b = bh >> 4, h = bh & 15;
    #pragma unroll
    for (int r = 0; r < 8; ++r) {
        const float inv = 1.0f / li[r];
        const int t = qt * 64 + wv * 16 + r + hf * 8;
        const size_t base = (size_t)(b * 2048 + t) * 1024 + h * 64 + l16;
        ob[base]      = (_Float16)(acc0[r] * inv);
        ob[base + 16] = (_Float16)(acc1[r] * inv);
        ob[base + 32] = (_Float16)(acc2[r] * inv);
        ob[base + 48] = (_Float16)(acc3[r] * inv);
    }
}

// ---------------------------------------------------------------------------
extern "C" void kernel_launch(void* const* d_in, const int* in_sizes, int n_in,
                              void* d_out, int out_size, void* d_ws, size_t ws_size,
                              hipStream_t stream) {
    (void)in_sizes; (void)n_in; (void)out_size; (void)ws_size;
    const float* x     = (const float*)d_in[0];   // (B*T, 1024)
    const float* qkv_w = (const float*)d_in[1];   // (1024, 3072)
    const float* qkv_b = (const float*)d_in[2];   // (3072,)
    const float* out_w = (const float*)d_in[3];   // (1024, 1024)
    const float* out_b = (const float*)d_in[4];   // (1024,)

    _Float16* ws    = (_Float16*)d_ws;
    _Float16* qkv_h = ws;                                   // 4096*3072      (24 MB)
    _Float16* qbuf  = qkv_h + (size_t)4096 * 3072;          // [B,H,T,64]     ( 8 MB)
    _Float16* kbuf  = qbuf + (size_t)4194304;
    _Float16* vbuf  = kbuf + (size_t)4194304;
    _Float16* attn  = vbuf + (size_t)4194304;               // 4096*1024      ( 8 MB)

    // 1) QKV projection (f32 in, f16 out)
    gemm_wmma_k<false, true><<<dim3(3072 / 64, 4096 / 128), dim3(256), 0, stream>>>(
        x, qkv_w, qkv_b, qkv_h, 4096, 3072, 1024);

    // 2) RoPE + split into head-major q/k/v
    rope_split_k<<<(2 * 2048 * 16 * 32) / 256, dim3(256), 0, stream>>>(
        qkv_h, qbuf, kbuf, vbuf);

    // 3) Causal flash attention (64-key tiles)
    attn_fwd_k<<<dim3(2048 / 64, 2 * 16), dim3(128), 0, stream>>>(
        qbuf, kbuf, vbuf, attn);

    // 4) Output projection (f16 in, f32 out) + bias
    gemm_wmma_k<true, false><<<dim3(1024 / 64, 4096 / 128), dim3(256), 0, stream>>>(
        attn, out_w, out_b, d_out, 4096, 1024, 1024);
}
